// GroupedQueryAttention_36558761624394
// MI455X (gfx1250) — matrix operand order repaired and checked by reading, not compiled, not run on hardware
//
#include <hip/hip_runtime.h>
#include <hip/hip_bf16.h>

// ---------------------------------------------------------------------------
// GroupedQueryAttention for MI455X (gfx1250, wave32, WMMA, async global->LDS)
//   B=2, S=2048, HIDDEN=2048, 16 q-heads, 4 kv-heads, head_dim=128
// Pipeline: cvt(fp32->bf16) -> WMMA GEMM (Q,K,V) -> flash GQA -> WMMA GEMM (O)
// Tiles double-buffered in LDS; K/A tiles moved with GLOBAL_LOAD_ASYNC_TO_LDS.
// ---------------------------------------------------------------------------

typedef __bf16 bf16;
typedef __attribute__((ext_vector_type(16))) __bf16 v16bf;
typedef __attribute__((ext_vector_type(8)))  __bf16 bf16x8;
typedef __attribute__((ext_vector_type(8)))  float  v8f;

#define HIDDEN   2048
#define NHEADS   16
#define NKV      4
#define HDIM     128
#define KVDIM    512            // NKV * HDIM
#define BATCH    2
#define SEQ      2048
#define MROWS    (BATCH * SEQ)  // 4096
#define QK_SCALE 0.08838834764831845f  // 128^-0.5

__device__ __forceinline__ bf16 f2bf(float f) {
    unsigned u = __builtin_bit_cast(unsigned, f);
    u += 0x7FFFu + ((u >> 16) & 1u);          // round-to-nearest-even
    unsigned short s = (unsigned short)(u >> 16);
    return __builtin_bit_cast(bf16, s);
}

__device__ __forceinline__ v8f vzero8() {
    v8f z;
#pragma unroll
    for (int i = 0; i < 8; ++i) z[i] = 0.f;
    return z;
}

// Async DMA: LDS[lds + IMM] = MEM[g + IMM], 16B per lane, tracked by ASYNCcnt.
template <int IMM>
__device__ __forceinline__ void async_b128(const bf16* g, const bf16* lds) {
    asm volatile("global_load_async_to_lds_b128 %0, %1, off offset:%2"
                 :: "v"((unsigned)(size_t)lds),
                    "v"((unsigned long long)g),
                    "i"(IMM)
                 : "memory");
}

__device__ __forceinline__ void wait_async0() {
#if __has_builtin(__builtin_amdgcn_s_wait_asynccnt)
    __builtin_amdgcn_s_wait_asynccnt(0);
#else
    asm volatile("s_wait_asynccnt 0x0" ::: "memory");
#endif
}

// ---------------------------------------------------------------------------
// fp32 -> bf16 conversion
// ---------------------------------------------------------------------------
__global__ void gqa_cvt_bf16(const float* __restrict__ src,
                             bf16* __restrict__ dst, int n) {
    int i = (blockIdx.x * blockDim.x + threadIdx.x) * 4;
    if (i + 4 <= n) {
        float4 f = *(const float4*)(src + i);
        dst[i + 0] = f2bf(f.x);
        dst[i + 1] = f2bf(f.y);
        dst[i + 2] = f2bf(f.z);
        dst[i + 3] = f2bf(f.w);
    } else {
        for (int j = i; j < n; ++j) dst[j] = f2bf(src[j]);
    }
}

// ---------------------------------------------------------------------------
// bf16 WMMA GEMM: C[M,N] = (A[M,K] @ B[K,N] + bias[N]) * out_scale
// 128x128 block tile, 8 waves (4x2), each wave 32x64 (2x4 of 16x16), K-step 32.
// A tile: async global->LDS.  B tile: transposed into LDS (coalesced global rd).
// Double-buffered; one barrier per K-step.
// ---------------------------------------------------------------------------
template <bool OUT_F32>
__global__ __launch_bounds__(256) void gqa_gemm_wmma(
    const bf16* __restrict__ A, const bf16* __restrict__ Bm,
    const float* __restrict__ bias, void* __restrict__ Cout,
    int M, int N, int K, float out_scale) {
    __shared__ bf16 a_lds[2][128 * 40];   // [row][k], stride 40
    __shared__ bf16 bT_lds[2][128 * 40];  // [n][k] transposed, stride 40

    const int t    = threadIdx.x;
    const int lane = t & 31;
    const int wid  = t >> 5;       // 0..7
    const int wm   = wid >> 1;     // 0..3 -> +32*wm rows
    const int wn   = wid & 1;      // 0..1 -> +64*wn cols
    const int half = lane >> 4;    // lane half (wave32 WMMA layout)
    const int l16  = lane & 15;

    const int blockN = blockIdx.x * 128;
    const int blockM = blockIdx.y * 128;

    auto issue_tile = [&](int k0, int buf) {
        {   // A tile 128x32 -> LDS via async DMA (2 threads/row, 32B each)
            int row = t >> 1, c = (t & 1) * 16;
            const bf16* g = A + (size_t)(blockM + row) * K + k0 + c;
            const bf16* d = &a_lds[buf][row * 40 + c];
            async_b128<0>(g, d);
            async_b128<16>(g, d);
        }
        {   // B tile 32x128 -> LDS transposed (8 threads/row-k, 16 n each)
            int kk = t >> 3, nb = (t & 7) * 16;
            const bf16* g = Bm + (size_t)(k0 + kk) * N + blockN + nb;
            bf16x8 v0 = *(const bf16x8*)g;
            bf16x8 v1 = *(const bf16x8*)(g + 8);
#pragma unroll
            for (int j = 0; j < 8; ++j) bT_lds[buf][(nb + j) * 40 + kk] = v0[j];
#pragma unroll
            for (int j = 0; j < 8; ++j) bT_lds[buf][(nb + 8 + j) * 40 + kk] = v1[j];
        }
    };

    v8f acc[2][4];
#pragma unroll
    for (int mt = 0; mt < 2; ++mt)
#pragma unroll
        for (int nt = 0; nt < 4; ++nt) acc[mt][nt] = vzero8();

    issue_tile(0, 0);
    wait_async0();
    __syncthreads();

    int buf = 0;
    for (int k0 = 0; k0 < K; k0 += 32) {
        if (k0 + 32 < K) {
            issue_tile(k0 + 32, buf ^ 1);   // overlap copy of t+1 with compute of t
            if (k0 + 64 < K) {              // prefetch t+2 into caches
                __builtin_prefetch(A + (size_t)(blockM + (t >> 1)) * K + k0 + 64, 0, 3);
                __builtin_prefetch(Bm + (size_t)(k0 + 64 + (t >> 3)) * N + blockN, 0, 3);
            }
        }

        // A fragments: a[i] = A[m][16*half + i]
        v16bf af[2];
#pragma unroll
        for (int mt = 0; mt < 2; ++mt) {
            const bf16* p = &a_lds[buf][(wm * 32 + mt * 16 + l16) * 40];
            bf16x8 lo = *(const bf16x8*)(p + 16 * half);
            bf16x8 hi = *(const bf16x8*)(p + 16 * half + 8);
#pragma unroll
            for (int i = 0; i < 8; ++i) { af[mt][i] = lo[i]; af[mt][i + 8] = hi[i]; }
        }
        // B fragments: b[i] = B[16*half + i][n]  (contiguous in bT_lds)
        v16bf bfr[4];
#pragma unroll
        for (int nt = 0; nt < 4; ++nt) {
            const bf16* p = &bT_lds[buf][(wn * 64 + nt * 16 + l16) * 40 + 16 * half];
            bf16x8 lo = *(const bf16x8*)p;
            bf16x8 hi = *(const bf16x8*)(p + 8);
#pragma unroll
            for (int i = 0; i < 8; ++i) { bfr[nt][i] = lo[i]; bfr[nt][i + 8] = hi[i]; }
        }
#pragma unroll
        for (int mt = 0; mt < 2; ++mt)
#pragma unroll
            for (int nt = 0; nt < 4; ++nt)
                acc[mt][nt] = __builtin_amdgcn_wmma_f32_16x16x32_bf16(
                    false, af[mt], false, bfr[nt], (short)0, acc[mt][nt], false, false);

        wait_async0();
        __syncthreads();
        buf ^= 1;
    }

    // epilogue: D layout -> memory (VGPR r -> row r + 8*half, lane -> col)
#pragma unroll
    for (int mt = 0; mt < 2; ++mt) {
#pragma unroll
        for (int nt = 0; nt < 4; ++nt) {
            int col = blockN + wn * 64 + nt * 16 + l16;
            float bv = bias[col];
#pragma unroll
            for (int r = 0; r < 8; ++r) {
                int row = blockM + wm * 32 + mt * 16 + r + 8 * half;
                float val = (acc[mt][nt][r] + bv) * out_scale;
                if (OUT_F32)
                    ((float*)Cout)[(size_t)row * N + col] = val;
                else
                    ((bf16*)Cout)[(size_t)row * N + col] = f2bf(val);
            }
        }
    }
}

// ---------------------------------------------------------------------------
// Flash grouped-query attention (1/sqrt(d) pre-folded into Q).
// Grid: (S/64, NHEADS, B); 128 threads = 4 waves; wave w owns 16 q rows.
// K tile: async global->LDS. V tile: transposed into LDS. Double-buffered.
// ---------------------------------------------------------------------------
__global__ __launch_bounds__(128) void gqa_flash_wmma(
    const bf16* __restrict__ Q, const bf16* __restrict__ Kc,
    const bf16* __restrict__ Vc, bf16* __restrict__ O) {
    __shared__ bf16 k_lds[2][32 * 136];   // [key][d], stride 136
    __shared__ bf16 vT_lds[2][128 * 40];  // [d][key], stride 40
    __shared__ bf16 p_lds[4][16 * 32];    // per-wave P tile [row][key]

    const int t    = threadIdx.x;
    const int lane = t & 31;
    const int w    = t >> 5;     // wave 0..3
    const int half = lane >> 4;
    const int l16  = lane & 15;

    const int b  = blockIdx.z;
    const int hq = blockIdx.y;   // query head 0..15
    const int h  = hq >> 2;      // kv head (GROUP=4)
    const int q0 = blockIdx.x * 64 + w * 16;

    auto issue_kv = [&](int kt, int buf) {
        {   // K tile 32x128 -> LDS via async DMA (4 threads/key, 64B each)
            int row = t >> 2, c = (t & 3) * 32;
            const bf16* g = Kc + ((size_t)(b * SEQ) + kt + row) * KVDIM + h * HDIM + c;
            const bf16* d = &k_lds[buf][row * 136 + c];
            async_b128<0>(g, d);
            async_b128<16>(g, d);
            async_b128<32>(g, d);
            async_b128<48>(g, d);
        }
        {   // V tile 32x128 -> LDS transposed
            int key = t & 31, dbase = (t >> 5) * 32;
            const bf16* g = Vc + ((size_t)(b * SEQ) + kt + key) * KVDIM + h * HDIM + dbase;
#pragma unroll
            for (int jj = 0; jj < 4; ++jj) {
                bf16x8 v = *(const bf16x8*)(g + jj * 8);
#pragma unroll
                for (int e = 0; e < 8; ++e)
                    vT_lds[buf][(dbase + jj * 8 + e) * 40 + key] = v[e];
            }
        }
    };

    // Q A-fragments for 4 d-chunks of 32 (rows q0..q0+15); scale pre-folded.
    v16bf qa[4];
    {
        const bf16* qrow = Q + ((size_t)(b * SEQ) + q0 + l16) * HIDDEN + hq * HDIM;
#pragma unroll
        for (int dc = 0; dc < 4; ++dc) {
            bf16x8 lo = *(const bf16x8*)(qrow + dc * 32 + 16 * half);
            bf16x8 hi = *(const bf16x8*)(qrow + dc * 32 + 16 * half + 8);
#pragma unroll
            for (int i = 0; i < 8; ++i) { qa[dc][i] = lo[i]; qa[dc][i + 8] = hi[i]; }
        }
    }

    float m_i[8], l_i[8];
    v8f accv[8];
#pragma unroll
    for (int r = 0; r < 8; ++r) { m_i[r] = -1e30f; l_i[r] = 0.f; }
#pragma unroll
    for (int j = 0; j < 8; ++j) accv[j] = vzero8();

    issue_kv(0, 0);
    wait_async0();
    __syncthreads();

    int buf = 0;
    for (int kt = 0; kt < SEQ; kt += 32) {
        if (kt + 32 < SEQ) issue_kv(kt + 32, buf ^ 1);  // overlap with compute

        // S = Q @ K^T : two 16x16 score tiles (keys 0..15, 16..31 of the tile)
        v8f s0 = vzero8(), s1 = vzero8();
#pragma unroll
        for (int dc = 0; dc < 4; ++dc) {
            v16bf kb0, kb1;
            const bf16* p0 = &k_lds[buf][(l16)*136      + dc * 32 + 16 * half];
            const bf16* p1 = &k_lds[buf][(16 + l16)*136 + dc * 32 + 16 * half];
            bf16x8 a0 = *(const bf16x8*)p0, a1 = *(const bf16x8*)(p0 + 8);
            bf16x8 c0 = *(const bf16x8*)p1, c1 = *(const bf16x8*)(p1 + 8);
#pragma unroll
            for (int i = 0; i < 8; ++i) {
                kb0[i] = a0[i]; kb0[i + 8] = a1[i];
                kb1[i] = c0[i]; kb1[i + 8] = c1[i];
            }
            s0 = __builtin_amdgcn_wmma_f32_16x16x32_bf16(false, qa[dc], false, kb0,
                                                         (short)0, s0, false, false);
            s1 = __builtin_amdgcn_wmma_f32_16x16x32_bf16(false, qa[dc], false, kb1,
                                                         (short)0, s1, false, false);
        }

        // online softmax per row (row = r + 8*half lives in one 16-lane half)
        float alpha[8];
#pragma unroll
        for (int r = 0; r < 8; ++r) {
            float x0 = s0[r], x1 = s1[r];
            float mx = fmaxf(x0, x1);
            mx = fmaxf(mx, __shfl_xor(mx, 1));
            mx = fmaxf(mx, __shfl_xor(mx, 2));
            mx = fmaxf(mx, __shfl_xor(mx, 4));
            mx = fmaxf(mx, __shfl_xor(mx, 8));
            float mnew = fmaxf(m_i[r], mx);
            float ep0 = __expf(x0 - mnew);
            float ep1 = __expf(x1 - mnew);
            float rs = ep0 + ep1;
            rs += __shfl_xor(rs, 1);
            rs += __shfl_xor(rs, 2);
            rs += __shfl_xor(rs, 4);
            rs += __shfl_xor(rs, 8);
            alpha[r] = __expf(m_i[r] - mnew);
            l_i[r] = l_i[r] * alpha[r] + rs;
            m_i[r] = mnew;
            s0[r] = ep0; s1[r] = ep1;
        }
        // stash P (bf16) for D->A re-layout (same-wave LDS, in-order), rescale acc
#pragma unroll
        for (int r = 0; r < 8; ++r) {
            p_lds[w][(r + 8 * half) * 32 + l16]      = f2bf(s0[r]);
            p_lds[w][(r + 8 * half) * 32 + 16 + l16] = f2bf(s1[r]);
        }
#pragma unroll
        for (int j = 0; j < 8; ++j)
#pragma unroll
            for (int r = 0; r < 8; ++r) accv[j][r] *= alpha[r];

        // P A-fragment (16x32 keys)
        v16bf pa;
        {
            const bf16* p = p_lds[w] + l16 * 32;
            bf16x8 lo = *(const bf16x8*)(p + 16 * half);
            bf16x8 hi = *(const bf16x8*)(p + 16 * half + 8);
#pragma unroll
            for (int i = 0; i < 8; ++i) { pa[i] = lo[i]; pa[i + 8] = hi[i]; }
        }
        // acc[j] += P @ V   (8 d-chunks of 16; b[i] = V[16*half+i][16j+n])
#pragma unroll
        for (int j = 0; j < 8; ++j) {
            const bf16* p = &vT_lds[buf][(16 * j + l16) * 40 + 16 * half];
            bf16x8 lo = *(const bf16x8*)p;
            bf16x8 hi = *(const bf16x8*)(p + 8);
            v16bf vb;
#pragma unroll
            for (int i = 0; i < 8; ++i) { vb[i] = lo[i]; vb[i + 8] = hi[i]; }
            accv[j] = __builtin_amdgcn_wmma_f32_16x16x32_bf16(false, pa, false, vb,
                                                              (short)0, accv[j],
                                                              false, false);
        }

        wait_async0();
        __syncthreads();
        buf ^= 1;
    }

    // epilogue: out = acc / l, write bf16 into [B,S,HIDDEN] at head hq
#pragma unroll
    for (int j = 0; j < 8; ++j) {
#pragma unroll
        for (int r = 0; r < 8; ++r) {
            int row = q0 + r + 8 * half;
            float val = accv[j][r] / l_i[r];
            O[((size_t)(b * SEQ) + row) * HIDDEN + hq * HDIM + 16 * j + l16] = f2bf(val);
        }
    }
}

// ---------------------------------------------------------------------------
// Host launcher
// ---------------------------------------------------------------------------
extern "C" void kernel_launch(void* const* d_in, const int* in_sizes, int n_in,
                              void* d_out, int out_size, void* d_ws, size_t ws_size,
                              hipStream_t stream) {
    const float* x  = (const float*)d_in[0];
    const float* Wq = (const float*)d_in[1];
    const float* bq = (const float*)d_in[2];
    const float* Wk = (const float*)d_in[3];
    const float* bk = (const float*)d_in[4];
    const float* Wv = (const float*)d_in[5];
    const float* bv = (const float*)d_in[6];
    const float* Wo = (const float*)d_in[7];
    const float* bo = (const float*)d_in[8];

    const int n_x  = MROWS * HIDDEN;   // 8388608
    const int n_wq = HIDDEN * HIDDEN;  // 4194304
    const int n_wk = HIDDEN * KVDIM;   // 1048576
    const int n_q  = MROWS * HIDDEN;
    const int n_k  = MROWS * KVDIM;

    char* ws = (char*)d_ws;
    size_t off = 0;
    bf16* x_bf  = (bf16*)(ws + off); off += (size_t)n_x  * 2;
    bf16* wq_bf = (bf16*)(ws + off); off += (size_t)n_wq * 2;
    bf16* wk_bf = (bf16*)(ws + off); off += (size_t)n_wk * 2;
    bf16* wv_bf = (bf16*)(ws + off); off += (size_t)n_wk * 2;
    bf16* wo_bf = (bf16*)(ws + off); off += (size_t)n_wq * 2;
    bf16* q_bf  = (bf16*)(ws + off); off += (size_t)n_q  * 2;
    bf16* k_bf  = (bf16*)(ws + off); off += (size_t)n_k  * 2;
    bf16* v_bf  = (bf16*)(ws + off); off += (size_t)n_k  * 2;
    bf16* a_bf  = (bf16*)(ws + off); off += (size_t)n_q  * 2;   // attn out (pre-Wo)

    auto cgrid = [](int n) { return (n / 4 + 255) / 256; };
    gqa_cvt_bf16<<<cgrid(n_x),  256, 0, stream>>>(x,  x_bf,  n_x);
    gqa_cvt_bf16<<<cgrid(n_wq), 256, 0, stream>>>(Wq, wq_bf, n_wq);
    gqa_cvt_bf16<<<cgrid(n_wk), 256, 0, stream>>>(Wk, wk_bf, n_wk);
    gqa_cvt_bf16<<<cgrid(n_wk), 256, 0, stream>>>(Wv, wv_bf, n_wk);
    gqa_cvt_bf16<<<cgrid(n_wq), 256, 0, stream>>>(Wo, wo_bf, n_wq);

    dim3 gq(HIDDEN / 128, MROWS / 128);   // (16,32)
    dim3 gk(KVDIM / 128,  MROWS / 128);   // (4,32)
    gqa_gemm_wmma<false><<<gq, 256, 0, stream>>>(x_bf, wq_bf, bq, q_bf,
                                                 MROWS, HIDDEN, HIDDEN, QK_SCALE);
    gqa_gemm_wmma<false><<<gk, 256, 0, stream>>>(x_bf, wk_bf, bk, k_bf,
                                                 MROWS, KVDIM, HIDDEN, 1.0f);
    gqa_gemm_wmma<false><<<gk, 256, 0, stream>>>(x_bf, wv_bf, bv, v_bf,
                                                 MROWS, KVDIM, HIDDEN, 1.0f);

    dim3 ga(SEQ / 64, NHEADS, BATCH);     // (32,16,2)
    gqa_flash_wmma<<<ga, 128, 0, stream>>>(q_bf, k_bf, v_bf, a_bf);

    gqa_gemm_wmma<true><<<gq, 256, 0, stream>>>(a_bf, wo_bf, bo, d_out,
                                                MROWS, HIDDEN, HIDDEN, 1.0f);
}
